// DiscontinuedGRU_65025804862175
// MI455X (gfx1250) — matrix operand order repaired and checked by reading, not compiled, not run on hardware
//
#include <hip/hip_runtime.h>

#define SEQL 2048
#define BATCH 64
#define INF 256
#define HID 256
#define MATELEMS (768 * 256)                       // elems per weight matrix
#define GXSTRIDE ((size_t)8192 * 16 * 3 * 256)     // per-direction gx elems

typedef __attribute__((ext_vector_type(16))) __bf16 v16bf;
typedef __attribute__((ext_vector_type(8)))  float  v8f;
typedef __attribute__((ext_vector_type(8)))  unsigned short v8u16;

__device__ __forceinline__ unsigned short f2bf(float f) {
    union { __bf16 b; unsigned short u; } v;
    v.b = (__bf16)f;                               // hardware cvt, RNE
    return v.u;
}
__device__ __forceinline__ float bf2f(unsigned short u) {
    union { unsigned u; float f; } v; v.u = ((unsigned)u) << 16; return v.f;
}

// ---------------------------------------------------------------------------
// Prepack weights f32 (3H,H) -> bf16 WMMA B-operand layout, kc-major:
// index = ((kc*48 + nt)*512) + lane*16 + slot ;  lane = n + 16*((kchunk/8)%2),
// slot = kchunk%8 + 8*(kchunk/16), tile nt = N-cols nt*16.., kc = K-chunk of 32.
// ---------------------------------------------------------------------------
__global__ void gru_prepack(const float* __restrict__ wihf,
                            const float* __restrict__ whhf,
                            const float* __restrict__ wihb,
                            const float* __restrict__ whhb,
                            unsigned short* __restrict__ wpack) {
    const int mi = blockIdx.y;                      // 0..3
    const int r  = blockIdx.x * 256 + threadIdx.x;  // 0..196607
    const int slot = r & 15;
    const int lane = (r >> 4) & 31;
    const int rem  = r >> 9;                        // 0..383
    const int nt   = rem % 48;
    const int kc   = rem / 48;
    const int k = (slot & 7) + 8 * (lane >> 4) + 16 * (slot >> 3);
    const int n = lane & 15;
    const float* W = (mi == 0) ? wihf : (mi == 1) ? whhf : (mi == 2) ? wihb : whhb;
    wpack[(size_t)mi * MATELEMS + r] = f2bf(W[(size_t)(nt * 16 + n) * 256 + kc * 32 + k]);
}

// ---------------------------------------------------------------------------
// Fully parallel input projection: gx = X @ Wih^T + bih (both dirs), bf16 out,
// packed as [t*4+bt][j(16)][g(3)][lane(32)][i(8)].  Grid 8192 x 256 threads.
// Two N-tiles processed together per wave for load/WMMA overlap.
// ---------------------------------------------------------------------------
__global__ __launch_bounds__(256)
void gru_gx(const float* __restrict__ X,
            const unsigned short* __restrict__ wpack,
            const float* __restrict__ bih_f, const float* __restrict__ bih_b,
            unsigned short* __restrict__ gx) {
    __shared__ unsigned short xpack[8 * 512];
    const int blk = blockIdx.x;                     // t*4 + bt
    const int t = blk >> 2, bt = blk & 3;
    const int tid = threadIdx.x;
    const int w = tid >> 5;                         // wave 0..7
    const int lane = tid & 31;
    const int nl = lane & 15;

    {   // cooperative pack of x tile into WMMA A layout
        const int m = tid >> 4;
        const int kq = tid & 15;
        const float* xrow = X + ((size_t)t * BATCH + bt * 16 + m) * INF + kq * 16;
        #pragma unroll
        for (int h = 0; h < 2; ++h) {
            const int qk = kq * 2 + h;
            v8u16 xv;
            #pragma unroll
            for (int i = 0; i < 8; ++i) xv[i] = f2bf(xrow[h * 8 + i]);
            *(v8u16*)&xpack[(qk >> 2) * 512 + (m + 16 * (qk & 1)) * 16 + ((qk >> 1) & 1) * 8] = xv;
        }
    }
    __syncthreads();

    v16bf a[8];
    #pragma unroll
    for (int kc = 0; kc < 8; ++kc) a[kc] = *(const v16bf*)&xpack[kc * 512 + lane * 16];

    for (int d = 0; d < 2; ++d) {
        const unsigned short* wih = wpack + (size_t)(2 * d) * MATELEMS;
        const float* bih = d ? bih_b : bih_f;
        unsigned short* gout = gx + (size_t)d * GXSTRIDE;
        for (int j3 = 0; j3 < 3; ++j3) {
            const int nt0 = __builtin_amdgcn_readfirstlane(w * 6 + j3 * 2);
            const unsigned short* wt0 = wih + (size_t)nt0 * 512;       // tile nt0
            const unsigned short* wt1 = wt0 + 512;                     // tile nt0+1
            const float bias0 = bih[nt0 * 16 + nl];
            const float bias1 = bih[nt0 * 16 + 16 + nl];
            v8f acc0, acc1;
            #pragma unroll
            for (int i = 0; i < 8; ++i) { acc0[i] = bias0; acc1[i] = bias1; }
            v16bf b0c = *(const v16bf*)(wt0 + lane * 16);
            v16bf b1c = *(const v16bf*)(wt1 + lane * 16);
            #pragma unroll
            for (int kc = 0; kc < 8; ++kc) {
                v16bf b0n, b1n;
                if (kc < 7) {
                    b0n = *(const v16bf*)(wt0 + (size_t)(kc + 1) * 24576 + lane * 16);
                    b1n = *(const v16bf*)(wt1 + (size_t)(kc + 1) * 24576 + lane * 16);
                }
                acc0 = __builtin_amdgcn_wmma_f32_16x16x32_bf16(false, a[kc], false, b0c, (short)0, acc0, false, false);
                acc1 = __builtin_amdgcn_wmma_f32_16x16x32_bf16(false, a[kc], false, b1c, (short)0, acc1, false, false);
                b0c = b0n; b1c = b1n;
            }
            #pragma unroll
            for (int p = 0; p < 2; ++p) {
                const int nt = nt0 + p;
                const int g = nt >> 4, j = nt & 15;
                const v8f acc = p ? acc1 : acc0;
                v8u16 ov;
                #pragma unroll
                for (int i = 0; i < 8; ++i) ov[i] = f2bf(acc[i]);
                __builtin_nontemporal_store(ov,
                    (v8u16*)(gout + ((((size_t)blk * 16 + j) * 3 + g) * 256 + lane * 8)));
            }
        }
    }
}

// ---------------------------------------------------------------------------
// Persistent recurrence. Grid = 8 (dir x batch-tile), 512 threads = 16 waves.
// PRE_GX: Whh fully on-chip: kc 0..2 in VGPRs, kc 3..7 in a 240 KB LDS stash.
// Fallback: Whh kc 0..4 in LDS, kc 5..7 + all of Wih streamed from L2.
// ---------------------------------------------------------------------------
template <bool PRE_GX>
__global__ __launch_bounds__(512, 1)
void gru_persistent(const float* __restrict__ X, const int* __restrict__ D,
                    const unsigned short* __restrict__ wpack,
                    const unsigned short* __restrict__ gx,
                    const float* __restrict__ bih_f, const float* __restrict__ bhh_f,
                    const float* __restrict__ bih_b, const float* __restrict__ bhh_b,
                    float* __restrict__ out) {
    constexpr int KREG   = PRE_GX ? 3 : 0;          // k-chunks held in VGPRs
    constexpr int KSTASH = 5;                       // k-chunks held in LDS
    extern __shared__ __align__(16) unsigned short dynsmem[];
    unsigned short* stash = dynsmem;                            // KSTASH*48KB
    unsigned short* xpack = dynsmem + KSTASH * 48 * 512;        // fallback only
    __shared__ unsigned short hpack[8 * 512];
    __shared__ float hf32[16 * 256];
    __shared__ float dmask[16];

    const int dir = blockIdx.x >> 2;
    const int bt  = blockIdx.x & 3;
    const int tid = threadIdx.x;
    const int w    = tid >> 5;
    const int w_s  = __builtin_amdgcn_readfirstlane(w);   // scalar wave id
    const int lane = tid & 31;
    const int nl   = lane & 15;
    const int lh   = lane >> 4;

    const float* bih = dir ? bih_b : bih_f;
    const float* bhh = dir ? bhh_b : bhh_f;
    const int col = w * 16 + nl;
    const float b_r  = (PRE_GX ? 0.f : bih[col])         + bhh[col];
    const float b_z  = (PRE_GX ? 0.f : bih[HID + col])   + bhh[HID + col];
    const float b_xn =  PRE_GX ? 0.f : bih[2 * HID + col];
    const float b_hn = bhh[2 * HID + col];

    const unsigned short* wih = wpack + (size_t)(2 * dir)     * MATELEMS;
    const unsigned short* whh = wpack + (size_t)(2 * dir + 1) * MATELEMS;
    const unsigned short* gxd = gx + (size_t)dir * GXSTRIDE;

    // register-resident Whh k-chunks (PRE_GX only)
    v16bf whhreg[KREG ? KREG : 1][3];
    if (PRE_GX) {
        #pragma unroll
        for (int kc = 0; kc < KREG; ++kc)
            #pragma unroll
            for (int g = 0; g < 3; ++g)
                whhreg[kc][g] = *(const v16bf*)(whh + (size_t)(kc * 24576 + (g * 16 + w_s) * 512 + lane * 16));
    }
    {   // copy Whh k-chunks [KREG, KREG+KSTASH) into LDS
        const float4* src = (const float4*)(whh + (size_t)KREG * 24576);
        float4* dst = (float4*)stash;
        const int n16 = KSTASH * 48 * 512 / 8;
        for (int i = tid; i < n16; i += 512) dst[i] = src[i];
    }

    const int cm = tid >> 5;
    const int qk = tid & 31;
    const int K0 = qk * 8;
    const int pdst = (qk >> 2) * 512 + (cm + 16 * (qk & 1)) * 16 + ((qk >> 1) & 1) * 8;

    for (int i = tid; i < 16 * 256; i += 512) hf32[i] = 0.f;
    v8f hprev;
    #pragma unroll
    for (int i = 0; i < 8; ++i) hprev[i] = 0.f;

    for (int s = 0; s < SEQL; ++s) {
        const int t = dir ? (SEQL - 1 - s) : s;
        __syncthreads();

        // launder streaming bases so LICM cannot hoist (and then spill) them
        size_t zg = 0;  asm volatile("" : "+s"(zg));
        unsigned zl = 0; asm volatile("" : "+v"(zl));

        {   // cooperative pack: masked h (and x in fallback) -> WMMA A layout
            v8u16 hv;
            #pragma unroll
            for (int i = 0; i < 8; ++i) hv[i] = f2bf(hf32[cm * 256 + K0 + i]);
            *(v8u16*)&hpack[pdst] = hv;
            if (!PRE_GX) {
                const float* xrow = X + ((size_t)t * BATCH + bt * 16 + cm) * INF + K0;
                v8u16 xv;
                #pragma unroll
                for (int i = 0; i < 8; ++i) xv[i] = f2bf(xrow[i]);
                *(v8u16*)&xpack[pdst] = xv;
                const int tn2 = dir ? (t - 1) : (t + 1);
                if (tn2 >= 0 && tn2 < SEQL)      // prefetch next step's x rows
                    __builtin_prefetch(X + ((size_t)tn2 * BATCH + bt * 16 + cm) * INF + K0, 0, 0);
            }
            if (tid < 16) {
                const int tn = dir ? (t - 1) : (t + 1);
                const int mi = (t > tn) ? t : tn;
                float keep = 1.f;
                if (mi < SEQL) keep = 1.f - (float)D[(size_t)mi * BATCH + bt * 16 + tid];
                dmask[tid] = keep;
            }
        }
        __syncthreads();

        v8f ar, az, axn, ahn;
        if (PRE_GX) {
            const unsigned short* gt = gxd + ((((size_t)t * 4 + bt) * 16 + w_s) * 3) * 256 + lane * 8;
            const v8u16 gr = *(const v8u16*)(gt);
            const v8u16 gz = *(const v8u16*)(gt + 256);
            const v8u16 gn = *(const v8u16*)(gt + 512);
            const int tp = dir ? (t - 1) : (t + 1);
            if (tp >= 0 && tp < SEQL) {          // prefetch next step's gx lines
                const unsigned short* gp = gxd + ((((size_t)tp * 4 + bt) * 16 + w_s) * 3) * 256 + lane * 8;
                __builtin_prefetch(gp, 0, 0);
                __builtin_prefetch(gp + 256, 0, 0);
                __builtin_prefetch(gp + 512, 0, 0);
            }
            #pragma unroll
            for (int i = 0; i < 8; ++i) {
                ar[i] = bf2f(gr[i]) + b_r;  az[i] = bf2f(gz[i]) + b_z;
                axn[i] = bf2f(gn[i]);       ahn[i] = b_hn;
            }
        } else {
            #pragma unroll
            for (int i = 0; i < 8; ++i) { ar[i] = b_r; az[i] = b_z; axn[i] = b_xn; ahn[i] = b_hn; }
        }

        #pragma unroll
        for (int kc = 0; kc < 8; ++kc) {
            const v16bf ah = *(const v16bf*)&hpack[kc * 512 + lane * 16];
            v16bf cr, cz, cn;
            if (kc < KREG) {
                cr = whhreg[kc][0]; cz = whhreg[kc][1]; cn = whhreg[kc][2];
            } else if (kc < KREG + KSTASH) {
                const unsigned sb = (unsigned)((kc - KREG) * 24576 + w_s * 512 + lane * 16) + zl;
                cr = *(const v16bf*)&stash[sb];
                cz = *(const v16bf*)&stash[sb + 16 * 512];
                cn = *(const v16bf*)&stash[sb + 32 * 512];
            } else {
                const size_t gb2 = (size_t)(kc * 24576 + w_s * 512 + lane * 16) + zg;
                cr = *(const v16bf*)(whh + gb2);
                cz = *(const v16bf*)(whh + gb2 + 16 * 512);
                cn = *(const v16bf*)(whh + gb2 + 32 * 512);
            }
            if (!PRE_GX) {
                const v16bf ax = *(const v16bf*)&xpack[(unsigned)(kc * 512 + lane * 16) + zl];
                const size_t wb = (size_t)(kc * 24576 + w_s * 512 + lane * 16) + zg;
                const v16bf br = *(const v16bf*)(wih + wb);
                const v16bf bz = *(const v16bf*)(wih + wb + 16 * 512);
                const v16bf bn = *(const v16bf*)(wih + wb + 32 * 512);
                ar  = __builtin_amdgcn_wmma_f32_16x16x32_bf16(false, ax, false, br, (short)0, ar,  false, false);
                az  = __builtin_amdgcn_wmma_f32_16x16x32_bf16(false, ax, false, bz, (short)0, az,  false, false);
                axn = __builtin_amdgcn_wmma_f32_16x16x32_bf16(false, ax, false, bn, (short)0, axn, false, false);
            }
            ar  = __builtin_amdgcn_wmma_f32_16x16x32_bf16(false, ah, false, cr, (short)0, ar,  false, false);
            az  = __builtin_amdgcn_wmma_f32_16x16x32_bf16(false, ah, false, cz, (short)0, az,  false, false);
            ahn = __builtin_amdgcn_wmma_f32_16x16x32_bf16(false, ah, false, cn, (short)0, ahn, false, false);
        }

        const size_t obase = ((size_t)t * BATCH + bt * 16) * 512 + (size_t)dir * 256 + col;
        #pragma unroll
        for (int i = 0; i < 8; ++i) {
            const float r = 1.f / (1.f + __expf(-ar[i]));
            const float z = 1.f / (1.f + __expf(-az[i]));
            const float nx = axn[i] + r * ahn[i];
            const float e = __expf(-2.f * nx);
            const float n = (1.f - e) / (1.f + e);      // tanh
            const float h = (1.f - z) * n + z * hprev[i];
            const int m = i + 8 * lh;
            __builtin_nontemporal_store(h, &out[obase + (size_t)m * 512]);
            const float hin = dmask[m] * h;
            hprev[i] = hin;
            hf32[m * 256 + col] = hin;
        }
    }
}

extern "C" void kernel_launch(void* const* d_in, const int* in_sizes, int n_in,
                              void* d_out, int out_size, void* d_ws, size_t ws_size,
                              hipStream_t stream) {
    (void)in_sizes; (void)n_in; (void)out_size;
    const float* X     = (const float*)d_in[0];
    const int*   D     = (const int*)  d_in[1];
    const float* Wih_f = (const float*)d_in[2];
    const float* Whh_f = (const float*)d_in[3];
    const float* bih_f = (const float*)d_in[4];
    const float* bhh_f = (const float*)d_in[5];
    const float* Wih_b = (const float*)d_in[6];
    const float* Whh_b = (const float*)d_in[7];
    const float* bih_b = (const float*)d_in[8];
    const float* bhh_b = (const float*)d_in[9];
    float* out = (float*)d_out;

    unsigned short* wpack = (unsigned short*)d_ws;                 // 1.5 MB
    const size_t gxoff = (size_t)2 << 20;
    unsigned short* gxp = (unsigned short*)((char*)d_ws + gxoff);  // 403 MB
    const size_t need = gxoff + 2 * GXSTRIDE * sizeof(unsigned short);

    gru_prepack<<<dim3(768, 4), 256, 0, stream>>>(Wih_f, Whh_f, Wih_b, Whh_b, wpack);

    if (ws_size >= need) {
        gru_gx<<<8192, 256, 0, stream>>>(X, wpack, bih_f, bih_b, gxp);
        const int dyn = 5 * 48 * 1024;                             // 240 KB
        hipFuncSetAttribute(reinterpret_cast<const void*>(gru_persistent<true>),
                            hipFuncAttributeMaxDynamicSharedMemorySize, dyn);
        gru_persistent<true><<<8, 512, dyn, stream>>>(X, D, wpack, gxp,
                                                      bih_f, bhh_f, bih_b, bhh_b, out);
    } else {
        const int dyn = 5 * 48 * 1024 + 8192;                      // 248 KB
        hipFuncSetAttribute(reinterpret_cast<const void*>(gru_persistent<false>),
                            hipFuncAttributeMaxDynamicSharedMemorySize, dyn);
        gru_persistent<false><<<8, 512, dyn, stream>>>(X, D, wpack, gxp,
                                                       bih_f, bhh_f, bih_b, bhh_b, out);
    }
}
